// Attention_14542759265074
// MI455X (gfx1250) — matrix operand order repaired and checked by reading, not compiled, not run on hardware
//
#include <hip/hip_runtime.h>

// Problem dimensions (match reference)
#define B_    4
#define LEN_  512
#define LEN1_ 512
#define DIN_  512
#define DH_   512
#define DM_   256
#define DV_   512

typedef __attribute__((ext_vector_type(16))) _Float16 v16h;
typedef __attribute__((ext_vector_type(8)))  _Float16 v8h;
typedef __attribute__((ext_vector_type(8)))  float    v8f;

// ---------------------------------------------------------------------------
// f32 -> f16 convert
// ---------------------------------------------------------------------------
__global__ void cvt_f32_f16(const float* __restrict__ s, _Float16* __restrict__ d, int n) {
    int i = blockIdx.x * blockDim.x + threadIdx.x;
    if (i < n) d[i] = (_Float16)s[i];
}

// values [B][L][V] f32 -> [B][V][L] f16 (so out-GEMM B-operand is K-contiguous)
__global__ void cvt_transpose_f16(const float* __restrict__ s, _Float16* __restrict__ d) {
    int i = blockIdx.x * blockDim.x + threadIdx.x;     // over B*L*V
    int v = i & (DV_ - 1);
    int l = (i >> 9) & (LEN_ - 1);
    int b = i >> 18;
    d[((size_t)b * DV_ + v) * LEN_ + l] = (_Float16)s[i];
}

// ---------------------------------------------------------------------------
// WMMA GEMM:  C[M][N] = A[M][K] * W[N][K]^T (+bias[N]), f16 in / f32 out.
// One wave per 16x16 C tile. blockIdx.y = batch (strides may be 0).
// Operand layouts per CDNA5 ISA 7.12.2 (16-bit A 16x32, B 32x16, f32 C/D).
// ---------------------------------------------------------------------------
__global__ void wmma_gemm_nt(const _Float16* __restrict__ A,
                             const _Float16* __restrict__ W,
                             const float*    __restrict__ bias,
                             float*          __restrict__ C,
                             int M, int N, int K,
                             long long aBatch, long long wBatch, long long cBatch) {
    const int bb = blockIdx.y;
    const _Float16* Ab = A + (size_t)bb * aBatch;
    const _Float16* Wb = W + (size_t)bb * wBatch;
    float*          Cb = C + (size_t)bb * cBatch;

    const int wavesPerBlk = blockDim.x >> 5;
    const int wave   = blockIdx.x * wavesPerBlk + (threadIdx.x >> 5);
    const int tilesN = N >> 4;
    const int m0 = (wave / tilesN) << 4;
    const int n0 = (wave % tilesN) << 4;
    const int lane = threadIdx.x & 31;
    const int hi   = lane >> 4;    // lane half (16..31)
    const int row  = lane & 15;    // A-row index / B,C column index

    const _Float16* arow = Ab + (size_t)(m0 + row) * K;
    const _Float16* wrow = Wb + (size_t)(n0 + row) * K;

    v8f acc = {};
    for (int k0 = 0; k0 < K; k0 += 32) {
        // A: a[e] -> K = k0 + 16*hi + e, contiguous 32B
        v16h a = *(const v16h*)(arow + k0 + 16 * hi);
        // B: b[e] -> K = k0 + 16*hi + e  (N = row), contiguous 32B
        v16h bm = *(const v16h*)(wrow + k0 + 16 * hi);
        acc = __builtin_amdgcn_wmma_f32_16x16x32_f16(
                  false, a, false, bm, (short)0, acc, false, false);
    }
    const float bv = bias ? bias[n0 + row] : 0.0f;
    #pragma unroll
    for (int e = 0; e < 8; ++e)
        Cb[(size_t)(m0 + 8 * hi + e) * N + n0 + row] = acc[e] + bv;
}

// ---------------------------------------------------------------------------
// Fused scores: scores[b][l][q] = sum_m w[m] * tanh(kx[b][l][m] + qh[b][q][m])
// Hardware v_tanh_f32, batched x4 to cover TRANS co-exec hazard.
// ---------------------------------------------------------------------------
__device__ __forceinline__ void tanh4(float x0, float x1, float x2, float x3,
                                      float& t0, float& t1, float& t2, float& t3) {
    asm volatile("v_tanh_f32 %0, %4\n\t"
                 "v_tanh_f32 %1, %5\n\t"
                 "v_tanh_f32 %2, %6\n\t"
                 "v_tanh_f32 %3, %7\n\t"
                 "v_nop"
                 : "=&v"(t0), "=&v"(t1), "=&v"(t2), "=&v"(t3)
                 : "v"(x0), "v"(x1), "v"(x2), "v"(x3));
}

#define PITCH 260   // 1040B row pitch: 16B-aligned, conflict-free across 16 rows

__global__ void scores_tanh(const float* __restrict__ kx, const float* __restrict__ qh,
                            const float* __restrict__ w, float* __restrict__ scores) {
    __shared__ float skx[16][PITCH];
    __shared__ float sqh[16][PITCH];
    __shared__ float sw[DM_];

    const int b  = blockIdx.z;
    const int l0 = blockIdx.y << 4;
    const int q0 = blockIdx.x << 4;
    const int tid = threadIdx.x;

    const float4* kxt = (const float4*)(kx + ((size_t)b * LEN_  + l0) * DM_);
    const float4* qht = (const float4*)(qh + ((size_t)b * LEN1_ + q0) * DM_);
    #pragma unroll
    for (int i = 0; i < 4; ++i) {
        int idx = tid + 256 * i;         // 1024 float4 = 16 rows x 256 floats
        int r = idx >> 6;
        int c = (idx & 63) << 2;
        *(float4*)&skx[r][c] = kxt[idx];
        *(float4*)&sqh[r][c] = qht[idx];
    }
    sw[tid] = w[tid];
    __syncthreads();

    const int qi = tid & 15, li = tid >> 4;
    const float* kr = skx[li];
    const float* qr = sqh[qi];
    float s = 0.0f;
    for (int m = 0; m < DM_; m += 4) {
        float4 kv = *(const float4*)(kr + m);
        float4 qv = *(const float4*)(qr + m);
        float4 wv = *(const float4*)(sw + m);
        float t0, t1, t2, t3;
        tanh4(kv.x + qv.x, kv.y + qv.y, kv.z + qv.z, kv.w + qv.w, t0, t1, t2, t3);
        s = fmaf(wv.x, t0, s);
        s = fmaf(wv.y, t1, s);
        s = fmaf(wv.z, t2, s);
        s = fmaf(wv.w, t3, s);
    }
    scores[((size_t)b * LEN_ + l0 + li) * LEN1_ + q0 + qi] = s;
}

// ---------------------------------------------------------------------------
// Column softmax over l (axis=1) + transposed f16 store: eT[b][q][l]
// ---------------------------------------------------------------------------
__global__ void softmax_transpose(const float* __restrict__ scores,
                                  _Float16* __restrict__ eT) {
    int idx = blockIdx.x * blockDim.x + threadIdx.x;   // 0..B*LEN1-1
    int b = idx >> 9, q = idx & (LEN1_ - 1);
    const float* col = scores + (size_t)b * LEN_ * LEN1_ + q;
    _Float16* dst = eT + ((size_t)b * LEN1_ + q) * LEN_;

    float mx = -3.4e38f;
    for (int l = 0; l < LEN_; ++l) mx = fmaxf(mx, col[(size_t)l * LEN1_]);
    float sum = 0.0f;
    for (int l = 0; l < LEN_; ++l) sum += __expf(col[(size_t)l * LEN1_] - mx);
    float inv = 1.0f / sum;
    for (int l = 0; l < LEN_; ++l)
        dst[l] = (_Float16)(__expf(col[(size_t)l * LEN1_] - mx) * inv);
}

// ---------------------------------------------------------------------------
extern "C" void kernel_launch(void* const* d_in, const int* in_sizes, int n_in,
                              void* d_out, int out_size, void* d_ws, size_t ws_size,
                              hipStream_t stream) {
    const float* query  = (const float*)d_in[0];   // [B,LEN1,DH]
    const float* keys   = (const float*)d_in[1];   // [B,LEN,DIN]
    const float* values = (const float*)d_in[2];   // [B,LEN,DV]
    const float* Wx     = (const float*)d_in[3];   // [DM,DIN]
    const float* Wh     = (const float*)d_in[4];   // [DM,DH]
    const float* bh     = (const float*)d_in[5];   // [DM]
    const float* w      = (const float*)d_in[6];   // [DM]
    float* out = (float*)d_out;                    // [B,LEN1,DV]

    char* ws = (char*)d_ws;
    size_t off = 0;
    auto carve = [&](size_t bytes) -> void* {
        void* p = ws + off;
        off = (off + bytes + 255) & ~(size_t)255;
        return p;
    };
    _Float16* keys16  = (_Float16*)carve((size_t)B_ * LEN_  * DIN_ * 2);
    _Float16* query16 = (_Float16*)carve((size_t)B_ * LEN1_ * DH_  * 2);
    _Float16* valT16  = (_Float16*)carve((size_t)B_ * DV_   * LEN_ * 2);
    _Float16* Wx16    = (_Float16*)carve((size_t)DM_ * DIN_ * 2);
    _Float16* Wh16    = (_Float16*)carve((size_t)DM_ * DH_  * 2);
    float*    kx      = (float*)carve((size_t)B_ * LEN_  * DM_ * 4);
    float*    qh      = (float*)carve((size_t)B_ * LEN1_ * DM_ * 4);
    float*    scores  = (float*)carve((size_t)B_ * LEN_ * LEN1_ * 4);
    _Float16* eT16    = (_Float16*)carve((size_t)B_ * LEN1_ * LEN_ * 2);

    const int nBig = B_ * LEN_ * DIN_;     // 1048576
    const int nW   = DM_ * DIN_;           // 131072

    cvt_f32_f16<<<(nBig + 255) / 256, 256, 0, stream>>>(keys,  keys16,  nBig);
    cvt_f32_f16<<<(nBig + 255) / 256, 256, 0, stream>>>(query, query16, nBig);
    cvt_f32_f16<<<(nW   + 255) / 256, 256, 0, stream>>>(Wx, Wx16, nW);
    cvt_f32_f16<<<(nW   + 255) / 256, 256, 0, stream>>>(Wh, Wh16, nW);
    cvt_transpose_f16<<<(nBig + 255) / 256, 256, 0, stream>>>(values, valT16);

    // kx = keys·Wxᵀ : M=B*LEN=2048, N=DM=256, K=512 -> 2048 tiles, 4 waves/blk
    wmma_gemm_nt<<<dim3(512, 1), 128, 0, stream>>>(
        keys16, Wx16, nullptr, kx, B_ * LEN_, DM_, DIN_, 0, 0, 0);
    // qh = query·Whᵀ + bh
    wmma_gemm_nt<<<dim3(512, 1), 128, 0, stream>>>(
        query16, Wh16, bh, qh, B_ * LEN1_, DM_, DH_, 0, 0, 0);

    // fused tanh + dot(w) — never materialize g
    scores_tanh<<<dim3(LEN1_ / 16, LEN_ / 16, B_), 256, 0, stream>>>(kx, qh, w, scores);

    // softmax over key axis, store eᵀ as f16
    softmax_transpose<<<(B_ * LEN1_) / 256, 256, 0, stream>>>(scores, eT16);

    // out[b] = eᵀ[b] (Q×L) · V[b] (L×V) : per-batch 512x512x512, batched via grid.y
    wmma_gemm_nt<<<dim3(1024 / 4, B_), 128, 0, stream>>>(
        eT16, valT16, nullptr, out, LEN1_, DV_, LEN_,
        (long long)LEN1_ * LEN_, (long long)DV_ * LEN_, (long long)LEN1_ * DV_);
}
